// AttModel_36180804501692
// MI455X (gfx1250) — matrix operand order repaired and checked by reading, not compiled, not run on hardware
//
#include <hip/hip_runtime.h>
#include <hip/hip_bf16.h>
#include <math.h>

typedef _Float16 h16;
typedef __attribute__((ext_vector_type(16))) _Float16 v16h;
typedef __attribute__((ext_vector_type(8)))  _Float16 v8h;
typedef __attribute__((ext_vector_type(8)))  float    v8f;

#define BNS   0.9999950000374997f   /* 1/sqrt(1+1e-5) */
#define PI_F  3.14159265358979f

// ---------------------------------------------------------------------------
// Register-blocked WMMA GEMM core, NT layout:
//   C[MT*16][NT*16] += A[MT*16][K] * B[NT*16][K]^T
// f16 in, f32 accumulate. MT=NT=2: per K=32 step 8x b128 loads feed 4
// v_wmma_f32_16x16x32_f16 (2x better WMMA:VMEM than 1x1) while staying well
// inside the VGPR budget (32 acc + ~64 fragments incl. pipelining).
// Fragment ABI (CDNA5 ISA 7.12.2):
//   A: lane L holds row M=L%16, K groups {0..7,16..23} (L<16) / {8..15,24..31}
//   B: lane L holds col N=L%16, K groups {0..7,16..23} (L<16) / {8..15,24..31}
// All lda/ldb multiples of 8 halves => 16B-aligned b128 loads.
// ---------------------------------------------------------------------------
template<int MT, int NT>
__device__ __forceinline__ void wmma_mma(const h16* __restrict__ A, int lda,
                                         const h16* __restrict__ B, int ldb, int K,
                                         v8f acc[MT][NT]) {
  const int lane = threadIdx.x & 31;
  const int r    = lane & 15;
  const int hi   = lane >> 4;
  const h16* ap = A + (long)r * lda + hi * 8;
  const h16* bp = B + (long)r * ldb + hi * 8;
  for (int k = 0; k < K; k += 32) {
    v16h av[MT], bv[NT];
#pragma unroll
    for (int i = 0; i < MT; ++i) {
      v8h a0 = *(const v8h*)(ap + (long)i * 16 * lda + k);
      v8h a1 = *(const v8h*)(ap + (long)i * 16 * lda + k + 16);
#pragma unroll
      for (int e = 0; e < 8; ++e) { av[i][e] = a0[e]; av[i][8+e] = a1[e]; }
    }
#pragma unroll
    for (int j = 0; j < NT; ++j) {
      v8h b0 = *(const v8h*)(bp + (long)j * 16 * ldb + k);
      v8h b1 = *(const v8h*)(bp + (long)j * 16 * ldb + k + 16);
#pragma unroll
      for (int e = 0; e < 8; ++e) { bv[j][e] = b0[e]; bv[j][8+e] = b1[e]; }
    }
#pragma unroll
    for (int i = 0; i < MT; ++i)
#pragma unroll
      for (int j = 0; j < NT; ++j)
        acc[i][j] = __builtin_amdgcn_wmma_f32_16x16x32_f16(
            false, av[i], false, bv[j], (short)0, acc[i][j], false, false);
  }
}
// C/D ABI: acc[i][j][r] is element (M = i*16 + r + 8*(lane>>4), N = j*16 + (lane&15)).

// ---------------------------------------------------------------------------
// conv1 (key & qry): out[512][P] = W1[512][960] * pinview[P][960]
// pin row stride 160 (135 ch + pad), k = t*160+c  =>  B row p = pin + p*160
// (im2col is a strided view). MT=2 (M=512 -> 16 groups).
// Epilogue: relu -> store f16 TRANSPOSED tmp[p][m] (stride 512); p>=Pvalid -> 0
// ---------------------------------------------------------------------------
template<int NT>
__global__ void __launch_bounds__(256, 1)
k_conv1(const h16* __restrict__ W, const h16* __restrict__ pin,
        h16* __restrict__ tmp, int baseRow, int nGroups, int Pvalid, int tmpRows) {
  int tid = blockIdx.x * (blockDim.x >> 5) + (threadIdx.x >> 5);
  int total = 256 * 16 * nGroups;
  if (tid >= total) return;
  int b = tid / (16 * nGroups);
  int rem = tid % (16 * nGroups);
  int mg = rem / nGroups, ng = rem % nGroups;
  const h16* A = W + (long)mg * 32 * 960;
  const h16* B = pin + (long)b * 136 * 160 + (long)(baseRow + ng * NT * 16) * 160;
  v8f z = {};
  v8f acc[2][NT];
#pragma unroll
  for (int i = 0; i < 2; ++i)
#pragma unroll
    for (int j = 0; j < NT; ++j) acc[i][j] = z;
  wmma_mma<2, NT>(A, 960, B, 160, 960, acc);
  int lane = threadIdx.x & 31;
  int n = lane & 15, mo = (lane >> 4) * 8;
#pragma unroll
  for (int i = 0; i < 2; ++i)
#pragma unroll
    for (int j = 0; j < NT; ++j) {
      int p = ng * NT * 16 + j * 16 + n;
      int m = mg * 32 + i * 16 + mo;
      h16* o = tmp + (long)b * tmpRows * 512 + (long)p * 512 + m;
#pragma unroll
      for (int r = 0; r < 8; ++r)
        o[r] = (p < Pvalid) ? (h16)fmaxf(acc[i][j][r], 0.f) : (h16)0.f;
    }
}

// ---------------------------------------------------------------------------
// conv2 (key & qry): out[512][P] = W2[512][2560] * tmpview[P][2560]
// tmp row stride 512, k = t*512+c => B row p = tmp + p*512 (reads rows p..p+4).
// Epilogue: relu -> store f32 outF[m][p] (ld outLd) for p < Pvalid
// ---------------------------------------------------------------------------
template<int NT>
__global__ void __launch_bounds__(256, 1)
k_conv2(const h16* __restrict__ W2, const h16* __restrict__ tmp,
        float* __restrict__ outF, int nGroups, int Pvalid, int tmpRows, int outLd) {
  int tid = blockIdx.x * (blockDim.x >> 5) + (threadIdx.x >> 5);
  int total = 256 * 16 * nGroups;
  if (tid >= total) return;
  int b = tid / (16 * nGroups);
  int rem = tid % (16 * nGroups);
  int mg = rem / nGroups, ng = rem % nGroups;
  const h16* A = W2 + (long)mg * 32 * 2560;
  const h16* B = tmp + (long)b * tmpRows * 512 + (long)(ng * NT * 16) * 512;
  v8f z = {};
  v8f acc[2][NT];
#pragma unroll
  for (int i = 0; i < 2; ++i)
#pragma unroll
    for (int j = 0; j < NT; ++j) acc[i][j] = z;
  wmma_mma<2, NT>(A, 2560, B, 512, 2560, acc);
  int lane = threadIdx.x & 31;
  int n = lane & 15, mo = (lane >> 4) * 8;
#pragma unroll
  for (int i = 0; i < 2; ++i)
#pragma unroll
    for (int j = 0; j < NT; ++j) {
      int p = ng * NT * 16 + j * 16 + n;
      int m = mg * 32 + i * 16 + mo;
      if (p < Pvalid) {
        float* o = outF + (long)b * 512 * outLd + (long)m * outLd + p;
#pragma unroll
        for (int r = 0; r < 8; ++r) o[(long)r * outLd] = fmaxf(acc[i][j][r], 0.f);
      }
    }
}

// ---------------------------------------------------------------------------
// gconv GEMM1: ZT[n][m] = (A[135pad160][K] * Bw[N][K]^T)^T  (f16 out, ld 160)
// MT=2 (M=160 -> 5 groups of 32).
// ---------------------------------------------------------------------------
template<int NT>
__global__ void __launch_bounds__(256, 1)
k_gemmT1(const h16* __restrict__ Ab, long aBatch, int lda, int K,
         const h16* __restrict__ Bw, int nGroups,
         h16* __restrict__ ZT, long ztBatch) {
  int tid = blockIdx.x * (blockDim.x >> 5) + (threadIdx.x >> 5);
  int total = 256 * 5 * nGroups;
  if (tid >= total) return;
  int b = tid / (5 * nGroups);
  int rem = tid % (5 * nGroups);
  int mg = rem / nGroups, ng = rem % nGroups;
  const h16* A = Ab + (long)b * aBatch + (long)mg * 32 * lda;
  const h16* B = Bw + (long)ng * NT * 16 * K;
  v8f z = {};
  v8f acc[2][NT];
#pragma unroll
  for (int i = 0; i < 2; ++i)
#pragma unroll
    for (int j = 0; j < NT; ++j) acc[i][j] = z;
  wmma_mma<2, NT>(A, lda, B, K, K, acc);
  int lane = threadIdx.x & 31;
  int n = lane & 15, mo = (lane >> 4) * 8;
#pragma unroll
  for (int i = 0; i < 2; ++i)
#pragma unroll
    for (int j = 0; j < NT; ++j) {
      int m = mg * 32 + i * 16 + mo;
      int nn = ng * NT * 16 + j * 16 + n;
      h16* o = ZT + (long)b * ztBatch + (long)nn * 160;
#pragma unroll
      for (int r = 0; r < 8; ++r) if (m + r < 135) o[m + r] = (h16)acc[i][j][r];
    }
}

// ---------------------------------------------------------------------------
// gconv GEMM2: y = tanh( g*((att@Z + bias)*BNS) + beta ) [+ res]
// A = att16 [160][160] (pads zeroed), B = ZT [512][160]. MT=2, NT=2.
// ---------------------------------------------------------------------------
__global__ void __launch_bounds__(256, 1)
k_gemmT2(const h16* __restrict__ att16, const h16* __restrict__ ZT,
         long ztBatch, const float* __restrict__ bias,
         const float* __restrict__ g, const float* __restrict__ beta,
         const float* __restrict__ res, float* __restrict__ y32,
         h16* __restrict__ y16, int useRes) {
  int tid = blockIdx.x * (blockDim.x >> 5) + (threadIdx.x >> 5);
  int total = 256 * 5 * 16;
  if (tid >= total) return;
  int b = tid / 80;
  int rem = tid % 80;
  int mg = rem / 16, ng = rem % 16;
  const h16* A = att16 + (long)mg * 32 * 160;
  const h16* B = ZT + (long)b * ztBatch + (long)(ng * 32) * 160;
  v8f z = {};
  v8f acc[2][2];
#pragma unroll
  for (int i = 0; i < 2; ++i)
#pragma unroll
    for (int j = 0; j < 2; ++j) acc[i][j] = z;
  wmma_mma<2, 2>(A, 160, B, 160, 160, acc);
  int lane = threadIdx.x & 31;
  int n = lane & 15, mo = (lane >> 4) * 8;
  long yb   = (long)b * 135 * 512;
  long yb16 = (long)b * 160 * 512;
#pragma unroll
  for (int i = 0; i < 2; ++i)
#pragma unroll
    for (int j = 0; j < 2; ++j) {
      int m  = mg * 32 + i * 16 + mo;
      int nn = ng * 32 + j * 16 + n;
#pragma unroll
      for (int r = 0; r < 8; ++r) {
        int mm = m + r;
        if (mm < 135) {
          int idx = mm * 512 + nn;
          float v = acc[i][j][r] + bias[nn];
          v = g[idx] * (v * BNS) + beta[idx];
          v = tanhf(v);
          if (useRes) v += res[yb + idx];
          y32[yb + idx]   = v;
          y16[yb16 + idx] = (h16)v;
        }
      }
    }
}

// gc7 GEMM2 (final): yfin = att@Z + bias + x   (f32, N=68). MT=2, NT=1.
__global__ void __launch_bounds__(256, 1)
k_gemmT2fin(const h16* __restrict__ att16, const h16* __restrict__ ZT7,
            const float* __restrict__ bias, const float* __restrict__ x32,
            float* __restrict__ yfin) {
  int tid = blockIdx.x * (blockDim.x >> 5) + (threadIdx.x >> 5);
  int total = 256 * 5 * 5;
  if (tid >= total) return;
  int b = tid / 25;
  int rem = tid % 25;
  int mg = rem / 5, ng = rem % 5;
  const h16* A = att16 + (long)mg * 32 * 160;
  const h16* B = ZT7 + (long)b * 80 * 160 + (long)(ng * 16) * 160;
  v8f z = {};
  v8f acc[2][1];
  acc[0][0] = z; acc[1][0] = z;
  wmma_mma<2, 1>(A, 160, B, 160, 160, acc);
  int lane = threadIdx.x & 31;
  int n = lane & 15, mo = (lane >> 4) * 8;
#pragma unroll
  for (int i = 0; i < 2; ++i) {
    int m  = mg * 32 + i * 16 + mo;
    int nn = ng * 16 + n;
#pragma unroll
    for (int r = 0; r < 8; ++r) {
      int mm = m + r;
      if (mm < 135 && nn < 68) {
        int idx = mm * 68 + nn;
        yfin[(long)b * 9180 + idx] = acc[i][0][r] + bias[nn] + x32[(long)b * 9180 + idx];
      }
    }
  }
}

// ------------------------- packing / elementwise ---------------------------
#define GSL(total) for (long i = (long)blockIdx.x * blockDim.x + threadIdx.x; \
                        i < (total); i += (long)gridDim.x * blockDim.x)

__global__ void k_pack_w1(const float* __restrict__ w, h16* __restrict__ o) {
  GSL(512L * 960) { int m = i / 960, k = i % 960, t = k / 160, c = k % 160;
    o[i] = (h16)((c < 135) ? w[((long)m * 135 + c) * 6 + t] : 0.f); }
}
__global__ void k_pack_w2(const float* __restrict__ w, h16* __restrict__ o) {
  GSL(512L * 2560) { int m = i / 2560, k = i % 2560, t = k / 512, c = k % 512;
    o[i] = (h16)w[((long)m * 512 + c) * 5 + t]; }
}
__global__ void k_pack_wT(const float* __restrict__ w, h16* __restrict__ o,
                          int Npad, int N, int K, int Kpad) {
  GSL((long)Npad * Kpad) { int n = i / Kpad, k = i % Kpad;
    o[i] = (h16)((n < N && k < K) ? w[(long)k * N + n] : 0.f); }
}
__global__ void k_pack_att(const float* __restrict__ a, h16* __restrict__ o) {
  GSL(160L * 160) { int m = i / 160, k = i % 160;
    o[i] = (h16)((m < 135 && k < 135) ? a[m * 135 + k] : 0.f); }
}
__global__ void k_pack_pin(const float* __restrict__ poses, h16* __restrict__ pin) {
  GSL(256L * 136 * 160) { int b = i / (136 * 160); int rem = i % (136 * 160);
    int r = rem / 160, c = rem % 160;
    pin[i] = (h16)((c < 135) ? poses[((long)b * 144 + r) * 135 + c] * 1e-3f : 0.f); }
}

// attention: att[p] = (q.k_p + 1e-15) / sum_p(...)
__global__ void k_att(const float* __restrict__ keyf, const float* __restrict__ qryf,
                      float* __restrict__ att) {
  __shared__ float sh[128];
  int b = blockIdx.x, t = threadIdx.x;
  float s = 0.f;
  if (t < 87) {
    const float* kf = keyf + (long)b * 512 * 96 + t;
    const float* qf = qryf + (long)b * 512;
    for (int d = 0; d < 512; ++d) s += qf[d] * kf[(long)d * 96];
    s += 1e-15f;
  }
  sh[t] = (t < 87) ? s : 0.f;
  __syncthreads();
  for (int off = 64; off > 0; off >>= 1) { if (t < off) sh[t] += sh[t + off]; __syncthreads(); }
  if (t < 87) att[(long)b * 96 + t] = s / sh[0];
}

// wsum[b][t][f] = sum_v att[v] * poses[b][v+t][f]
__global__ void k_wsum(const float* __restrict__ poses, const float* __restrict__ att,
                       float* __restrict__ wsum) {
  GSL(256L * 34 * 135) {
    int b = i / (34 * 135); int rem = i % (34 * 135); int t = rem / 135, f = rem % 135;
    const float* a = att + (long)b * 96;
    const float* p = poses + ((long)b * 144 + t) * 135 + f;
    float s = 0.f;
    for (int v = 0; v < 87; ++v) s += a[v] * p[(long)v * 135];
    wsum[i] = s;
  }
}

// orthonormal DCT-II; inverse = transpose
__device__ __forceinline__ float dctv(int k, int t) {
  float w = sqrtf((k == 0 ? 1.f : 2.f) / 34.f);
  return w * cosf(PI_F * (t + 0.5f) * (float)k / 34.f);
}

// x[b][f][0:34]=DCT(padded last frames); x[b][f][34:68]=DCT(att-weighted window)
__global__ void k_x(const float* __restrict__ poses, const float* __restrict__ wsum,
                    float* __restrict__ x32, h16* __restrict__ x16) {
  GSL(256L * 135 * 68) {
    int b = i / (135 * 68); int rem = i % (135 * 68); int f = rem / 68, kc = rem % 68;
    float s = 0.f;
    if (kc < 34) {
      for (int t = 0; t < 34; ++t) {
        int row = (t < 10) ? 110 + t : 119;
        s += dctv(kc, t) * poses[((long)b * 144 + row) * 135 + f];
      }
    } else {
      int k = kc - 34;
      const float* wb = wsum + (long)b * 34 * 135 + f;
      for (int t = 0; t < 34; ++t) s += dctv(k, t) * wb[t * 135];
    }
    x32[i] = s;
    x16[(long)b * 160 * 96 + (long)f * 96 + kc] = (h16)s;
  }
}

// preds[b][t][f] = sum_k IDCT[10+t][k]*yfin[b][f][k],  IDCT[t][k]=DCT[k][t]
__global__ void k_out(const float* __restrict__ yfin, float* __restrict__ out) {
  GSL(256L * 24 * 135) {
    int b = i / (24 * 135); int rem = i % (24 * 135); int t = rem / 135, f = rem % 135;
    const float* y = yfin + (long)b * 9180 + (long)f * 68;
    float s = 0.f;
    for (int k = 0; k < 34; ++k) s += dctv(k, 10 + t) * y[k];
    out[i] = s;
  }
}

// ---------------------------------------------------------------------------
static inline int gb(long waves) { return (int)((waves + 7) / 8); }

extern "C" void kernel_launch(void* const* d_in, const int* in_sizes, int n_in,
                              void* d_out, int out_size, void* d_ws, size_t ws_size,
                              hipStream_t stream) {
  (void)in_sizes; (void)n_in; (void)out_size; (void)ws_size;
  const float* poses   = (const float*)d_in[0];
  const float* qw1     = (const float*)d_in[1];
  const float* qw2     = (const float*)d_in[2];
  const float* kw1     = (const float*)d_in[3];
  const float* kw2     = (const float*)d_in[4];
  const float* gc1_att = (const float*)d_in[5];
  const float* gc1_w   = (const float*)d_in[6];
  const float* gc1_b   = (const float*)d_in[7];
  const float* bn1_g   = (const float*)d_in[8];
  const float* bn1_b   = (const float*)d_in[9];
  const float* gcb_att = (const float*)d_in[10];
  const float* gcb_w   = (const float*)d_in[11];
  const float* gcb_b   = (const float*)d_in[12];
  const float* gcb_g   = (const float*)d_in[13];
  const float* gcb_beta= (const float*)d_in[14];
  const float* gc7_att = (const float*)d_in[15];
  const float* gc7_w   = (const float*)d_in[16];
  const float* gc7_b   = (const float*)d_in[17];
  float* outp = (float*)d_out;

  size_t off = 0;
  char* base = (char*)d_ws;
  auto alloc = [&](size_t bytes) -> char* {
    char* p = base + off; off += (bytes + 255) & ~(size_t)255; return p; };

  h16*  W1k  = (h16*) alloc(512L*960*2);
  h16*  W1q  = (h16*) alloc(512L*960*2);
  h16*  W2k  = (h16*) alloc(512L*2560*2);
  h16*  W2q  = (h16*) alloc(512L*2560*2);
  h16*  B1   = (h16*) alloc(512L*96*2);
  h16*  Bw   = (h16*) alloc(4L*512*512*2);
  h16*  B7   = (h16*) alloc(80L*512*2);
  h16*  att1 = (h16*) alloc(160L*160*2);
  h16*  attb = (h16*) alloc(4L*160*160*2);
  h16*  att7 = (h16*) alloc(160L*160*2);
  h16*  pin  = (h16*) alloc(256L*136*160*2);
  h16*  tmpk = (h16*) alloc(256L*100*512*2);
  h16*  tmpq = (h16*) alloc(256L*20*512*2);
  float* keyf= (float*)alloc(256L*512*96*4);
  float* qryf= (float*)alloc(256L*512*4);
  float* attw= (float*)alloc(256L*96*4);
  float* wsum= (float*)alloc(256L*34*135*4);
  float* x32 = (float*)alloc(256L*135*68*4);
  h16*  x16  = (h16*) alloc(256L*160*96*2);
  h16*  ZT   = (h16*) alloc(256L*512*160*2);
  h16*  ZT7  = (h16*) alloc(256L*80*160*2);
  float* yA32= (float*)alloc(256L*135*512*4);
  h16*  yA16 = (h16*) alloc(256L*160*512*2);
  float* yB32= (float*)alloc(256L*135*512*4);
  h16*  yB16 = (h16*) alloc(256L*160*512*2);
  float* yfin= (float*)alloc(256L*135*68*4);

  // zero padded regions (graph-capture-safe stream memsets)
  hipMemsetAsync(tmpk, 0, 256L*100*512*2, stream);
  hipMemsetAsync(tmpq, 0, 256L*20*512*2, stream);
  hipMemsetAsync(x16,  0, 256L*160*96*2, stream);
  hipMemsetAsync(ZT,   0, 256L*512*160*2, stream);
  hipMemsetAsync(ZT7,  0, 256L*80*160*2, stream);
  hipMemsetAsync(yA16, 0, 256L*160*512*2, stream);
  hipMemsetAsync(yB16, 0, 256L*160*512*2, stream);

  // pack weights / attention matrices / input
  k_pack_w1 <<<512, 256, 0, stream>>>(kw1, W1k);
  k_pack_w1 <<<512, 256, 0, stream>>>(qw1, W1q);
  k_pack_w2 <<<1024,256, 0, stream>>>(kw2, W2k);
  k_pack_w2 <<<1024,256, 0, stream>>>(qw2, W2q);
  k_pack_wT <<<256, 256, 0, stream>>>(gc1_w, B1, 512, 512, 68, 96);
  for (int l = 0; l < 4; ++l)
    k_pack_wT<<<1024,256,0, stream>>>(gcb_w + (long)l*512*512, Bw + (long)l*512*512,
                                      512, 512, 512, 512);
  k_pack_wT <<<256, 256, 0, stream>>>(gc7_w, B7, 80, 68, 512, 512);
  k_pack_att<<<100, 256, 0, stream>>>(gc1_att, att1);
  for (int l = 0; l < 4; ++l)
    k_pack_att<<<100,256,0, stream>>>(gcb_att + (long)l*135*135, attb + (long)l*160*160);
  k_pack_att<<<100, 256, 0, stream>>>(gc7_att, att7);
  k_pack_pin<<<2048,256, 0, stream>>>(poses, pin);

  // attention branch: conv1 -> conv2 -> score normalization
  k_conv1<2><<<gb(256L*16*3), 256, 0, stream>>>(W1k, pin, tmpk,   0, 3, 91, 100);
  k_conv1<1><<<gb(256L*16*1), 256, 0, stream>>>(W1q, pin, tmpq, 110, 1,  5,  20);
  k_conv2<2><<<gb(256L*16*3), 256, 0, stream>>>(W2k, tmpk, keyf, 3, 87, 100, 96);
  k_conv2<1><<<gb(256L*16*1), 256, 0, stream>>>(W2q, tmpq, qryf, 1,  1,  20,  1);
  k_att     <<<256, 128, 0, stream>>>(keyf, qryf, attw);

  // build GCN input x (DCT of padded input ++ DCT of attention-weighted window)
  k_wsum <<<2048, 256, 0, stream>>>(poses, attw, wsum);
  k_x    <<<2048, 256, 0, stream>>>(poses, wsum, x32, x16);

  const long ztB = 512L*160, aB = 160L*512;
  // gc1
  k_gemmT1<2><<<gb(256L*5*16), 256, 0, stream>>>(x16, 160L*96, 96, 96, B1, 16, ZT, ztB);
  k_gemmT2   <<<gb(256L*5*16), 256, 0, stream>>>(att1, ZT, ztB, gc1_b, bn1_g, bn1_b,
                                                 yA32, yA32, yA16, 0);
  // 2 residual stages x 2 gconv layers
  for (int s = 0; s < 2; ++s) {
    int l0 = 2*s, l1 = 2*s + 1;
    k_gemmT1<2><<<gb(256L*5*16), 256, 0, stream>>>(yA16, aB, 512, 512,
                                                   Bw + (long)l0*512*512, 16, ZT, ztB);
    k_gemmT2   <<<gb(256L*5*16), 256, 0, stream>>>(attb + (long)l0*160*160, ZT, ztB,
                                                   gcb_b + l0*512, gcb_g + (long)l0*69120,
                                                   gcb_beta + (long)l0*69120,
                                                   yA32, yB32, yB16, 0);
    k_gemmT1<2><<<gb(256L*5*16), 256, 0, stream>>>(yB16, aB, 512, 512,
                                                   Bw + (long)l1*512*512, 16, ZT, ztB);
    k_gemmT2   <<<gb(256L*5*16), 256, 0, stream>>>(attb + (long)l1*160*160, ZT, ztB,
                                                   gcb_b + l1*512, gcb_g + (long)l1*69120,
                                                   gcb_beta + (long)l1*69120,
                                                   yA32, yA32, yA16, 1);
  }
  // gc7 + residual x, then IDCT of last 24 frames
  k_gemmT1<1><<<gb(256L*5*5), 256, 0, stream>>>(yA16, aB, 512, 512, B7, 5, ZT7, 80L*160);
  k_gemmT2fin<<<gb(256L*5*5), 256, 0, stream>>>(att7, ZT7, gc7_b, x32, yfin);
  k_out      <<<1024, 256, 0, stream>>>(yfin, outp);
}
